// Encoder_2817498546299
// MI455X (gfx1250) — matrix operand order repaired and checked by reading, not compiled, not run on hardware
//
#include <hip/hip_runtime.h>
#include <math.h>

// ---------------------------------------------------------------------------
// CDNA5 (gfx1250) bf16 WMMA encoder layer, v2.
// All WMMA operands arranged so the K-dimension is contiguous in memory:
// weights pre-transposed to [N][K] bf16, V pre-transposed to [B,H,64,S],
// position tables to [H,257,64]. B fragments become 2x global_load_b128.
// ---------------------------------------------------------------------------

typedef __attribute__((ext_vector_type(16))) __bf16 v16bf;
typedef __attribute__((ext_vector_type(8)))  float  v8f;

union BF16x16 {
    v16bf v;
    unsigned short h[16];
    unsigned int   u[8];
};
union F32x8 {
    v8f   v;
    float f[8];
};

__device__ __forceinline__ unsigned short f2bfbits(float f) {
    unsigned int u = __builtin_bit_cast(unsigned int, f);
    unsigned int r = u + 0x7fffu + ((u >> 16) & 1u);   // round-nearest-even
    return (unsigned short)(r >> 16);
}

#define WMMA_BF16(A, Bm, C) \
    __builtin_amdgcn_wmma_f32_16x16x32_bf16(false, (A), false, (Bm), (short)0, (C), false, false)

// ---------------------------------------------------------------------------
// LayerNorm over D per row -> bf16 output
// ---------------------------------------------------------------------------
__global__ __launch_bounds__(256)
void ln_kernel(const float* __restrict__ X, const float* __restrict__ g,
               const float* __restrict__ bta, unsigned short* __restrict__ out, int Dn) {
    __shared__ float r1[256], r2[256];
    const int row = blockIdx.x;
    const int tid = threadIdx.x;
    const float* xr = X + (size_t)row * Dn;
    float s = 0.f, s2 = 0.f;
    for (int i = tid; i < Dn; i += 256) { float v = xr[i]; s += v; s2 += v * v; }
    r1[tid] = s; r2[tid] = s2;
    __syncthreads();
    for (int o = 128; o > 0; o >>= 1) {
        if (tid < o) { r1[tid] += r1[tid + o]; r2[tid] += r2[tid + o]; }
        __syncthreads();
    }
    const float mean = r1[0] / (float)Dn;
    const float var  = r2[0] / (float)Dn - mean * mean;
    const float rstd = rsqrtf(var + 1e-8f);
    for (int i = tid; i < Dn; i += 256)
        out[(size_t)row * Dn + i] = f2bfbits((xr[i] - mean) * rstd * g[i] + bta[i]);
}

// ---------------------------------------------------------------------------
// Batched LDS-tiled transpose+convert: f32 in[z][rows][cols] -> bf16 out[z][cols][rows]
// ---------------------------------------------------------------------------
__global__ __launch_bounds__(256)
void wT_kernel(const float* __restrict__ in, unsigned short* __restrict__ out,
               int rows, int cols, long inBatch, long outBatch) {
    __shared__ unsigned short tile[64][65];
    const int r0 = blockIdx.y * 64, c0 = blockIdx.x * 64;
    const float* src = in + (size_t)blockIdx.z * inBatch;
    unsigned short* dst = out + (size_t)blockIdx.z * outBatch;
    for (int i = threadIdx.x; i < 4096; i += 256) {
        const int r = i >> 6, c = i & 63;
        const int gr = r0 + r, gc = c0 + c;
        tile[r][c] = (gr < rows && gc < cols)
                         ? f2bfbits(src[(size_t)gr * cols + gc]) : (unsigned short)0;
    }
    __syncthreads();
    for (int i = threadIdx.x; i < 4096; i += 256) {
        const int c = i >> 6, r = i & 63;
        const int gr = r0 + r, gc = c0 + c;
        if (gc < cols && gr < rows) dst[(size_t)gc * rows + gr] = tile[r][c];
    }
}

// ---------------------------------------------------------------------------
// V transpose: v_bf [B,S,D] -> vT [B*H, 64, S]   (per-head head_dim x seq)
// ---------------------------------------------------------------------------
__global__ __launch_bounds__(256)
void vT_kernel(const unsigned short* __restrict__ v_bf, unsigned short* __restrict__ vT,
               int S, int D, int Hn) {
    __shared__ unsigned short tile[64][65];
    const int bh = blockIdx.y;
    const int b = bh / Hn, h = bh % Hn;
    const int s0 = blockIdx.x * 64;
    const unsigned short* src = v_bf + (size_t)b * S * D + h * 64;
    unsigned short* dst = vT + (size_t)bh * 64 * S;
    for (int i = threadIdx.x; i < 4096; i += 256) {
        const int s = i >> 6, d = i & 63;
        tile[s][d] = src[(size_t)(s0 + s) * D + d];
    }
    __syncthreads();
    for (int i = threadIdx.x; i < 4096; i += 256) {
        const int d = i >> 6, s = i & 63;
        dst[(size_t)d * S + s0 + s] = tile[s][d];
    }
}

// ---------------------------------------------------------------------------
// GEMM: C[M,N] = A[M,K] @ BT[N,K]^T  (+bias)(+gelu)(+resid)
// block = 4 waves; wave computes 32x64 (2 M-tiles x 4 N-tiles, 8 WMMA/K-step).
// M multiple of 128; N,K multiples of 64.
// ---------------------------------------------------------------------------
__global__ __launch_bounds__(128)
void gemm_bf16(const unsigned short* __restrict__ A, const unsigned short* __restrict__ BT,
               float* outF, unsigned short* outB,
               const float* bias, const float* resid,
               int M, int N, int K, int doGelu) {
    const int lane = threadIdx.x & 31;
    const int wave = threadIdx.x >> 5;
    const int ln15 = lane & 15;
    const int half = lane >> 4;
    const int m0 = (blockIdx.y * 4 + wave) * 32;
    const int n0 = blockIdx.x * 64;
    if (m0 >= M) return;

    F32x8 acc[2][4];
#pragma unroll
    for (int mi = 0; mi < 2; ++mi)
#pragma unroll
        for (int t = 0; t < 4; ++t)
#pragma unroll
            for (int r = 0; r < 8; ++r) acc[mi][t].f[r] = 0.f;

    const unsigned int* Arow0 = (const unsigned int*)(A + (size_t)(m0 + ln15) * K);
    const unsigned int* Arow1 = (const unsigned int*)(A + (size_t)(m0 + 16 + ln15) * K);
    const unsigned int* Brow[4];
#pragma unroll
    for (int t = 0; t < 4; ++t)
        Brow[t] = (const unsigned int*)(BT + (size_t)(n0 + t * 16 + ln15) * K);

    for (int kk = 0; kk < K; kk += 32) {
        BF16x16 a0, a1;
        const int ia = (kk >> 1) + 8 * half;        // A pair v: K = kk + 16*half + 2v
#pragma unroll
        for (int v = 0; v < 4; ++v) { a0.u[v] = Arow0[ia + v];     a1.u[v] = Arow1[ia + v]; }
#pragma unroll
        for (int v = 0; v < 4; ++v) { a0.u[4 + v] = Arow0[ia + 4 + v]; a1.u[4 + v] = Arow1[ia + 4 + v]; }

        const int ib = (kk >> 1) + 8 * half;        // B pair v: K = kk + 16*half + 2v
#pragma unroll
        for (int t = 0; t < 4; ++t) {
            BF16x16 b;
#pragma unroll
            for (int v = 0; v < 8; ++v) b.u[v] = Brow[t][ib + v];
            acc[0][t].v = WMMA_BF16(a0.v, b.v, acc[0][t].v);
            acc[1][t].v = WMMA_BF16(a1.v, b.v, acc[1][t].v);
        }
    }

#pragma unroll
    for (int mi = 0; mi < 2; ++mi)
#pragma unroll
        for (int t = 0; t < 4; ++t) {
            const int n = n0 + t * 16 + ln15;
            const float bv = bias ? bias[n] : 0.f;
#pragma unroll
            for (int r = 0; r < 8; ++r) {
                const int m = m0 + mi * 16 + r + 8 * half;
                float val = acc[mi][t].f[r] + bv;
                if (doGelu) val = 0.5f * val * (1.0f + erff(val * 0.7071067811865475f));
                const size_t idx = (size_t)m * N + n;
                if (resid) val += resid[idx];
                if (outF) outF[idx] = val;
                if (outB) outB[idx] = f2bfbits(val);
            }
        }
}

// ---------------------------------------------------------------------------
// Batched position GEMM: Out[z,S,257] = Xh[b,:,h*64:+64] @ PosT[h]^T, z = b*H+h
// Xh bf16 [B,S,D]; PosT bf16 [H,257,64]; K=64 (2 steps); N guarded at 257.
// ---------------------------------------------------------------------------
__global__ __launch_bounds__(128)
void pos_gemm(const unsigned short* __restrict__ Xh, const unsigned short* __restrict__ PosT,
              float* __restrict__ Out, int S, int D, int Hn) {
    const int NP = 257;
    const int z = blockIdx.z;
    const int b = z / Hn, h = z % Hn;
    const int lane = threadIdx.x & 31;
    const int wave = threadIdx.x >> 5;
    const int ln15 = lane & 15;
    const int half = lane >> 4;
    const int m0 = (blockIdx.y * 4 + wave) * 16;
    const int n0 = blockIdx.x * 64;

    const unsigned short* Abase = Xh + (size_t)b * S * D + h * 64;
    const unsigned short* Bbase = PosT + (size_t)h * NP * 64;
    float* Cbase = Out + (size_t)z * S * NP;

    F32x8 acc[4];
#pragma unroll
    for (int t = 0; t < 4; ++t)
#pragma unroll
        for (int r = 0; r < 8; ++r) acc[t].f[r] = 0.f;

    const unsigned int* Arow = (const unsigned int*)(Abase + (size_t)(m0 + ln15) * D);

#pragma unroll
    for (int s = 0; s < 2; ++s) {                   // kk = 32*s
        BF16x16 a;
        const int ia = 16 * s + 8 * half;
#pragma unroll
        for (int v = 0; v < 4; ++v) a.u[v] = Arow[ia + v];
#pragma unroll
        for (int v = 0; v < 4; ++v) a.u[4 + v] = Arow[ia + 4 + v];

        const int ib = 16 * s + 8 * half;
#pragma unroll
        for (int t = 0; t < 4; ++t) {
            const int n = n0 + t * 16 + ln15;
            const bool nok = (n < NP);
            const unsigned int* Bp = (const unsigned int*)(Bbase + (size_t)(nok ? n : 0) * 64);
            BF16x16 bf;
#pragma unroll
            for (int v = 0; v < 8; ++v) bf.u[v] = nok ? Bp[ib + v] : 0u;
            acc[t].v = WMMA_BF16(a.v, bf.v, acc[t].v);
        }
    }

#pragma unroll
    for (int t = 0; t < 4; ++t) {
        const int n = n0 + t * 16 + ln15;
        if (n < NP) {
#pragma unroll
            for (int r = 0; r < 8; ++r)
                Cbase[(size_t)(m0 + r + 8 * half) * NP + n] = acc[t].f[r];
        }
    }
}

// ---------------------------------------------------------------------------
// Banded disentangled attention. One wave per (b, h, 16-query tile).
// Band 2W+1 = 257 keys -> 272-col window [q0-128, q0+143], padded to 288.
// P@V uses vT [B*H, 64, S] so B-fragment key pairs are one dword.
// ---------------------------------------------------------------------------
__global__ __launch_bounds__(32)
void attn_kernel(const unsigned short* __restrict__ q_bf, const unsigned short* __restrict__ k_bf,
                 const unsigned short* __restrict__ vT,
                 const float* __restrict__ c2p, const float* __restrict__ p2c,
                 const unsigned char* __restrict__ mask,
                 unsigned short* __restrict__ ctx_bf, int S, int D, int Hn) {
    const int SW = 128, NP = 257;
    __shared__ float          sc[16][272];
    __shared__ unsigned short P[16][288];

    const int b = blockIdx.y / Hn, h = blockIdx.y % Hn;
    const int q0 = blockIdx.x * 16;
    const int lane = threadIdx.x & 31;
    const int ln15 = lane & 15;
    const int half = lane >> 4;
    const float LV = -3.4028234e38f;

    // Q fragments for hd=64 (2 K-steps), reused for all 17 key tiles.
    BF16x16 aq[2];
    {
        const unsigned int* Arow =
            (const unsigned int*)(q_bf + (size_t)(b * S + q0 + ln15) * D + h * 64);
#pragma unroll
        for (int s = 0; s < 2; ++s) {
            const int ia = 16 * s + 8 * half;
#pragma unroll
            for (int v = 0; v < 4; ++v) aq[s].u[v] = Arow[ia + v];
#pragma unroll
            for (int v = 0; v < 4; ++v) aq[s].u[4 + v] = Arow[ia + 4 + v];
        }
    }

    // Phase A: scores = Q.K^T + c2p gather + p2c gather, masked -> LDS
    for (int t = 0; t < 17; ++t) {
        const int k0 = q0 - SW + t * 16;
        const int key = k0 + ln15;
        const bool kok = (key >= 0) && (key < S);
        F32x8 acc;
#pragma unroll
        for (int r = 0; r < 8; ++r) acc.f[r] = 0.f;

        if (k0 + 15 >= 0 && k0 < S) {               // wave-uniform
            const unsigned int* Kp =
                (const unsigned int*)(k_bf + (size_t)(b * S + (kok ? key : 0)) * D + h * 64 + 16 * half);
#pragma unroll
            for (int s = 0; s < 2; ++s) {
                BF16x16 bk;
#pragma unroll
                for (int v = 0; v < 8; ++v) bk.u[v] = kok ? Kp[16 * s + v] : 0u;
                acc.v = WMMA_BF16(aq[s].v, bk.v, acc.v);
            }
        }
#pragma unroll
        for (int r = 0; r < 8; ++r) {
            const int qi = q0 + r + 8 * half;
            const int d = key - qi;
            float val = LV;
            if (kok && d >= -SW && d <= SW && mask[(size_t)b * S + key] && mask[(size_t)b * S + qi]) {
                const size_t base = (size_t)(b * Hn + h) * S;
                val = acc.f[r] + c2p[(base + qi) * NP + (d + SW)]
                               + p2c[(base + key) * NP + (SW - d)];
            }
            sc[r + 8 * half][t * 16 + ln15] = val;
        }
    }
    __syncthreads();

    // Phase B: masked softmax per row; probabilities stored as bf16 in LDS.
    {
        const int row = ln15;
        const int cbeg = half * 136, cend = cbeg + 136;
        float mx = LV;
        for (int c = cbeg; c < cend; ++c) mx = fmaxf(mx, sc[row][c]);
        mx = fmaxf(mx, __shfl_xor(mx, 16, 32));
        float sum = 0.f;
        for (int c = cbeg; c < cend; ++c) {
            float e = expf(sc[row][c] - mx);
            sum += e;
            sc[row][c] = e;
        }
        sum += __shfl_xor(sum, 16, 32);
        const float inv = 1.0f / sum;
        for (int c = cbeg; c < cend; ++c) P[row][c] = f2bfbits(sc[row][c] * inv);
        if (half) {
            for (int c = 272; c < 288; ++c) P[row][c] = 0;
        }
    }
    __syncthreads();

    // Phase C: ctx = P @ V over padded 288 window (9 K-steps, 4 N-tiles).
    F32x8 o[4];
#pragma unroll
    for (int t = 0; t < 4; ++t)
#pragma unroll
        for (int r = 0; r < 8; ++r) o[t].f[r] = 0.f;

    const unsigned short* Vh = vT + (size_t)(b * Hn + h) * 64 * S;

    for (int kk = 0; kk < 288; kk += 32) {
        BF16x16 ap;
        const unsigned int* Pp = (const unsigned int*)&P[ln15][0];
        const int ia = (kk >> 1) + 8 * half;
#pragma unroll
        for (int v = 0; v < 4; ++v) ap.u[v] = Pp[ia + v];
#pragma unroll
        for (int v = 0; v < 4; ++v) ap.u[4 + v] = Pp[ia + 4 + v];

#pragma unroll
        for (int t = 0; t < 4; ++t) {
            const unsigned short* Vrow = Vh + (size_t)(t * 16 + ln15) * S;
            BF16x16 bv;
#pragma unroll
            for (int v = 0; v < 8; ++v) {
                // pair covers keys (key0, key0+1); key0 even so pair validity is atomic
                const int c0 = kk + 16 * half + 2 * v;
                const int key0 = q0 - SW + c0;
                bv.u[v] = (c0 < 272 && key0 >= 0 && key0 < S)
                              ? *(const unsigned int*)(Vrow + key0) : 0u;
            }
            o[t].v = WMMA_BF16(ap.v, bv.v, o[t].v);
        }
    }

#pragma unroll
    for (int t = 0; t < 4; ++t)
#pragma unroll
        for (int r = 0; r < 8; ++r)
            ctx_bf[(size_t)(b * S + q0 + r + 8 * half) * D + h * 64 + t * 16 + ln15] =
                f2bfbits(o[t].f[r]);
}

// ---------------------------------------------------------------------------
// Host launcher
// ---------------------------------------------------------------------------
extern "C" void kernel_launch(void* const* d_in, const int* in_sizes, int n_in,
                              void* d_out, int out_size, void* d_ws, size_t ws_size,
                              hipStream_t stream) {
    (void)in_sizes; (void)n_in; (void)out_size; (void)ws_size;
    const int Bn = 4, Sn = 1024, Dn = 1024, Hn = 16, In = 4096, NP = 257;
    const int M = Bn * Sn;

    const float* hidden = (const float*)d_in[0];
    const unsigned char* mask = (const unsigned char*)d_in[1];
    const float* ln1g = (const float*)d_in[2];
    const float* ln1b = (const float*)d_in[3];
    const float* wq   = (const float*)d_in[4];
    const float* wk   = (const float*)d_in[5];
    const float* wv   = (const float*)d_in[6];
    const float* posq = (const float*)d_in[7];
    const float* posk = (const float*)d_in[8];
    const float* wo   = (const float*)d_in[9];
    const float* bo   = (const float*)d_in[10];
    const float* ln2g = (const float*)d_in[11];
    const float* ln2b = (const float*)d_in[12];
    const float* win  = (const float*)d_in[13];
    const float* bin  = (const float*)d_in[14];
    const float* wout = (const float*)d_in[15];
    const float* bout = (const float*)d_in[16];
    float* out = (float*)d_out;

    char* ws = (char*)d_ws;
    size_t off = 0;
    auto take = [&](size_t bytes) {
        char* p = ws + off;
        off = (off + bytes + 255) & ~(size_t)255;
        return p;
    };
    unsigned short* x_bf    = (unsigned short*)take((size_t)M * Dn * 2);
    unsigned short* wqT     = (unsigned short*)take((size_t)Dn * Dn * 2);
    unsigned short* wkT     = (unsigned short*)take((size_t)Dn * Dn * 2);
    unsigned short* wvT     = (unsigned short*)take((size_t)Dn * Dn * 2);
    unsigned short* woT     = (unsigned short*)take((size_t)Dn * Dn * 2);
    unsigned short* winT    = (unsigned short*)take((size_t)Dn * In * 2);
    unsigned short* woutT   = (unsigned short*)take((size_t)In * Dn * 2);
    unsigned short* pqT     = (unsigned short*)take((size_t)Hn * 64 * NP * 2);
    unsigned short* pkT     = (unsigned short*)take((size_t)Hn * 64 * NP * 2);
    unsigned short* q_bf    = (unsigned short*)take((size_t)M * Dn * 2);
    unsigned short* k_bf    = (unsigned short*)take((size_t)M * Dn * 2);
    unsigned short* v_bf    = (unsigned short*)take((size_t)M * Dn * 2);
    unsigned short* vT      = (unsigned short*)take((size_t)M * Dn * 2);
    float*          c2p     = (float*)take((size_t)Bn * Hn * Sn * NP * 4);
    float*          p2c     = (float*)take((size_t)Bn * Hn * Sn * NP * 4);
    unsigned short* ctx_bf  = (unsigned short*)take((size_t)M * Dn * 2);
    float*          hbuf    = (float*)take((size_t)M * Dn * 4);
    unsigned short* y_bf    = (unsigned short*)take((size_t)M * Dn * 2);
    unsigned short* ff1_bf  = (unsigned short*)take((size_t)M * In * 2);

    // Weight / position-table transposes (f32 [rows,cols] -> bf16 [cols,rows])
    wT_kernel<<<dim3(Dn / 64, Dn / 64, 1), dim3(256), 0, stream>>>(wq, wqT, Dn, Dn, 0, 0);
    wT_kernel<<<dim3(Dn / 64, Dn / 64, 1), dim3(256), 0, stream>>>(wk, wkT, Dn, Dn, 0, 0);
    wT_kernel<<<dim3(Dn / 64, Dn / 64, 1), dim3(256), 0, stream>>>(wv, wvT, Dn, Dn, 0, 0);
    wT_kernel<<<dim3(Dn / 64, Dn / 64, 1), dim3(256), 0, stream>>>(wo, woT, Dn, Dn, 0, 0);
    wT_kernel<<<dim3(In / 64, Dn / 64, 1), dim3(256), 0, stream>>>(win, winT, Dn, In, 0, 0);
    wT_kernel<<<dim3(Dn / 64, In / 64, 1), dim3(256), 0, stream>>>(wout, woutT, In, Dn, 0, 0);
    wT_kernel<<<dim3((NP + 63) / 64, 1, Hn), dim3(256), 0, stream>>>(
        posq, pqT, 64, NP, (long)64 * NP, (long)NP * 64);
    wT_kernel<<<dim3((NP + 63) / 64, 1, Hn), dim3(256), 0, stream>>>(
        posk, pkT, 64, NP, (long)64 * NP, (long)NP * 64);

    // LN1 -> x_bf
    ln_kernel<<<dim3(M), dim3(256), 0, stream>>>(hidden, ln1g, ln1b, x_bf, Dn);

    // QKV projections (bf16 outputs)
    gemm_bf16<<<dim3(Dn / 64, M / 128), dim3(128), 0, stream>>>(
        x_bf, wqT, nullptr, q_bf, nullptr, nullptr, M, Dn, Dn, 0);
    gemm_bf16<<<dim3(Dn / 64, M / 128), dim3(128), 0, stream>>>(
        x_bf, wkT, nullptr, k_bf, nullptr, nullptr, M, Dn, Dn, 0);
    gemm_bf16<<<dim3(Dn / 64, M / 128), dim3(128), 0, stream>>>(
        x_bf, wvT, nullptr, v_bf, nullptr, nullptr, M, Dn, Dn, 0);

    // V transpose for the P@V phase
    vT_kernel<<<dim3(Sn / 64, Bn * Hn), dim3(256), 0, stream>>>(v_bf, vT, Sn, Dn, Hn);

    // Disentangled position tables: c2p = q@pos_key, p2c = k@pos_query
    pos_gemm<<<dim3((NP + 63) / 64, Sn / 64, Bn * Hn), dim3(128), 0, stream>>>(
        q_bf, pkT, c2p, Sn, Dn, Hn);
    pos_gemm<<<dim3((NP + 63) / 64, Sn / 64, Bn * Hn), dim3(128), 0, stream>>>(
        k_bf, pqT, p2c, Sn, Dn, Hn);

    // Banded attention -> ctx (bf16)
    attn_kernel<<<dim3(Sn / 16, Bn * Hn), dim3(32), 0, stream>>>(
        q_bf, k_bf, vT, c2p, p2c, mask, ctx_bf, Sn, Dn, Hn);

    // Output projection + residual -> h (f32)
    gemm_bf16<<<dim3(Dn / 64, M / 128), dim3(128), 0, stream>>>(
        ctx_bf, woT, hbuf, nullptr, bo, hidden, M, Dn, Dn, 0);

    // LN2 -> y_bf
    ln_kernel<<<dim3(M), dim3(256), 0, stream>>>(hbuf, ln2g, ln2b, y_bf, Dn);

    // FFN1: gelu(y @ w_in + b_in) -> bf16
    gemm_bf16<<<dim3(In / 64, M / 128), dim3(128), 0, stream>>>(
        y_bf, winT, nullptr, ff1_bf, bin, nullptr, M, In, Dn, 1);

    // FFN2: h + ff1 @ w_out + b_out -> d_out (f32)
    gemm_bf16<<<dim3(Dn / 64, M / 128), dim3(128), 0, stream>>>(
        ff1_bf, woutT, out, nullptr, bout, hbuf, M, Dn, In, 0);
}
